// PWCDCNet_11158325035492
// MI455X (gfx1250) — matrix operand order repaired and checked by reading, not compiled, not run on hardware
//
#include <hip/hip_runtime.h>

// ---------------------------------------------------------------------------
// CDNA5 (gfx1250) types
// ---------------------------------------------------------------------------
typedef __bf16 bf16_t;
typedef bf16_t v16bf __attribute__((ext_vector_type(16)));
typedef float  v8f   __attribute__((ext_vector_type(8)));

// Packed f32->bf16 RNE convert: one VALU op per pair (V_CVT_PK_BF16_F32).
__device__ __forceinline__ unsigned pack_bf2(float lo, float hi) {
  unsigned r;
  asm("v_cvt_pk_bf16_f32 %0, %1, %2" : "=v"(r) : "v"(lo), "v"(hi));
  return r;
}
__device__ __forceinline__ int imin(int a, int b) { return a < b ? a : b; }
__device__ __forceinline__ int imax(int a, int b) { return a > b ? a : b; }

// ---------------------------------------------------------------------------
// Implicit-GEMM 3x3 conv via v_wmma_f32_16x16x32_bf16, register-direct
// fragments (no LDS, no barriers, no divergent loads).
//  M = 16*MT output channels (blockIdx.y), N = 128 pixels (blockIdx.x, 8 waves),
//  K = Cin*9 in chunks of 32. MT accumulator tiles share one B fragment:
//  im2col gather cost amortized over MT WMMAs per iteration.
//  k order used (contiguous, same for A and B):
//   A (16x32): lane row = M; half 0 holds K 0..15, half 1 K 16..31
//              -> 16 consecutive K of one weight row.
//   B (32x16): lane col = N (pixel); half 0 holds K 0..15, half 1 K 16..31
//              -> 16 consecutive K of one im2col column.
//  OOB handled by clamped addresses + value selects (EXEC stays all-ones).
//  Fused: bias, leaky-relu(0.1), output scale, optional residual add.
// ---------------------------------------------------------------------------
#define TW 128
#define KC 32

template <int MT>
__global__ __launch_bounds__(256)
void conv3x3_wmma(const float* __restrict__ in, const float* __restrict__ wgt,
                  const float* __restrict__ bias, float* __restrict__ out,
                  const float* __restrict__ resid,
                  int Cin, int Cout, int Hin, int Win, int Hout, int Wout,
                  int stride, int pad, int dil, int relu, float oscale)
{
  const int t    = threadIdx.x;
  const int lane = t & 31;
  const int wv   = t >> 5;             // wave -> 16-pixel sub-tile
  const int half = lane >> 4;
  const int mr   = lane & 15;
  const int npix = Hout * Wout;
  const int pix0 = blockIdx.x * TW;
  const int co0  = blockIdx.y * (16 * MT);
  const int K    = Cin * 9;

  // ---- this lane's im2col column (pixel) ----
  const int p      = pix0 + wv * 16 + mr;
  const bool pval  = (p < npix);
  const int pc     = pval ? p : (npix - 1);
  const int oy     = pc / Wout;
  const int ox     = pc - oy * Wout;
  const int iy0    = oy * stride - pad;
  const int ix0    = ox * stride - pad;

  // ---- this lane's weight rows (one output channel per M-tile) ----
  const float* wrow[MT];
  bool cval[MT];
  #pragma unroll
  for (int mt = 0; mt < MT; ++mt) {
    const int c = co0 + 16 * mt + mr;
    cval[mt] = (c < Cout);
    wrow[mt] = wgt + (size_t)(cval[mt] ? c : (Cout - 1)) * K;
  }

  v8f acc[MT];
  #pragma unroll
  for (int mt = 0; mt < MT; ++mt)
    acc[mt] = (v8f){0.f, 0.f, 0.f, 0.f, 0.f, 0.f, 0.f, 0.f};

  for (int kb = 0; kb < K; kb += KC) {
    // ---- B fragment: 16 consecutive K of this lane's im2col column ----
    union { v16bf v; unsigned u[8]; } fb;
    {
      int kg = kb + half * 16;
      int ci = kg / 9;
      int r  = kg - ci * 9;            // incremental (ci, r) thereafter
      float bv[16];
      #pragma unroll
      for (int i = 0; i < 16; ++i) {
        const int ky = (r >= 6) ? 2 : ((r >= 3) ? 1 : 0);
        const int kx = r - 3 * ky;
        const int y  = iy0 + ky * dil;
        const int x  = ix0 + kx * dil;
        const bool inb = pval && (kg < K) &&
                         ((unsigned)y < (unsigned)Hin) &&
                         ((unsigned)x < (unsigned)Win);
        const int yc  = imin(imax(y, 0), Hin - 1);
        const int xc  = imin(imax(x, 0), Win - 1);
        const int cic = imin(ci, Cin - 1);
        const float v = in[((size_t)cic * Hin + yc) * Win + xc];
        bv[i] = inb ? v : 0.f;
        ++kg; ++r;
        if (r == 9) { r = 0; ++ci; }
      }
      #pragma unroll
      for (int vv = 0; vv < 8; ++vv)
        fb.u[vv] = pack_bf2(bv[2 * vv], bv[2 * vv + 1]);
    }

    // ---- A fragments (one per M-tile) + WMMA, all sharing fb ----
    #pragma unroll
    for (int mt = 0; mt < MT; ++mt) {
      union { v16bf v; unsigned u[8]; } fa;
      #pragma unroll
      for (int vv = 0; vv < 8; ++vv) {
        const int ka = half * 16 + 2 * vv;
        const int kg = kb + ka;
        float a0 = wrow[mt][imin(kg,     K - 1)];
        float a1 = wrow[mt][imin(kg + 1, K - 1)];
        a0 = (cval[mt] && kg     < K) ? a0 : 0.f;
        a1 = (cval[mt] && kg + 1 < K) ? a1 : 0.f;
        fa.u[vv] = pack_bf2(a0, a1);
      }
      acc[mt] = __builtin_amdgcn_wmma_f32_16x16x32_bf16(
          /*neg_a=*/false, fa.v, /*neg_b=*/false, fb.v,
          /*c_mod=*/(short)0, acc[mt], /*reuse_a=*/false, /*reuse_b=*/false);
    }
  }

  // ---- epilogue: C/D layout lanes0-15 M=r, lanes16-31 M=8+r ; N=lane&15 ----
  if (pval) {
    #pragma unroll
    for (int mt = 0; mt < MT; ++mt) {
      #pragma unroll
      for (int r = 0; r < 8; ++r) {
        const int m  = r + 8 * half;
        const int cc = co0 + 16 * mt + m;
        if (cc < Cout) {
          float v = acc[mt][r] + bias[cc];
          if (relu) v = (v > 0.f) ? v : 0.1f * v;
          v *= oscale;
          const size_t oi = (size_t)cc * npix + p;
          if (resid) v += resid[oi];
          out[oi] = v;
        }
      }
    }
  }
}

// ---------------------------------------------------------------------------
// Bilinear helpers (half-pixel centers, edge clamp ~ border padding)
// ---------------------------------------------------------------------------
__device__ __forceinline__ float bilerp_clamp(const float* __restrict__ im,
                                              int H, int W, float y, float x)
{
  y = fminf(fmaxf(y, 0.f), (float)(H - 1));
  x = fminf(fmaxf(x, 0.f), (float)(W - 1));
  const int y0 = (int)y, x0 = (int)x;
  const int y1 = imin(y0 + 1, H - 1), x1 = imin(x0 + 1, W - 1);
  const float wy = y - y0, wx = x - x0;
  const float a = im[y0 * W + x0], b = im[y0 * W + x1];
  const float c = im[y1 * W + x0], d = im[y1 * W + x1];
  return a * (1.f - wx) * (1.f - wy) + b * wx * (1.f - wy)
       + c * (1.f - wx) * wy + d * wx * wy;
}

// ---------------------------------------------------------------------------
// 81-way cost volume: warp f2 by (flow + search offset), mean |f1 - warp(f2)|.
// ---------------------------------------------------------------------------
__global__ void dense_warp_kernel(const float* __restrict__ fa,
                                  const float* __restrict__ fc,
                                  const float* __restrict__ si,
                                  const float* __restrict__ sj,
                                  const float* __restrict__ flow, int hf, int wf,
                                  float* __restrict__ out, int C, int H, int W)
{
  const int idx = blockIdx.x * blockDim.x + threadIdx.x;
  const int hw = H * W;
  if (idx >= 81 * hw) return;
  const int d = idx / hw;
  const int p = idx - d * hw;
  const int py = p / W, px = p - py * W;

  float fx = 0.f, fy = 0.f;
  if (flow) {
    const float sy = (py + 0.5f) * ((float)hf / (float)H) - 0.5f;
    const float sx = (px + 0.5f) * ((float)wf / (float)W) - 0.5f;
    fx = bilerp_clamp(flow,            hf, wf, sy, sx) * (2.f / (float)wf);
    fy = bilerp_clamp(flow + hf * wf,  hf, wf, sy, sx) * (2.f / (float)hf);
  }
  const float gx = -1.f + 2.f * px / (float)(W - 1) + fx + sj[d] / (float)W;
  const float gy = -1.f + 2.f * py / (float)(H - 1) + fy + si[d] / (float)H;

  float x = (gx + 1.f) * (W * 0.5f) - 0.5f;
  float y = (gy + 1.f) * (H * 0.5f) - 0.5f;
  x = fminf(fmaxf(x, 0.f), (float)(W - 1));
  y = fminf(fmaxf(y, 0.f), (float)(H - 1));
  const int x0 = (int)x, y0 = (int)y;
  const int x1 = imin(x0 + 1, W - 1), y1 = imin(y0 + 1, H - 1);
  const float wx = x - x0, wy = y - y0;
  const float w00 = (1.f - wx) * (1.f - wy), w01 = wx * (1.f - wy);
  const float w10 = (1.f - wx) * wy,         w11 = wx * wy;
  const int i00 = y0 * W + x0, i01 = y0 * W + x1;
  const int i10 = y1 * W + x0, i11 = y1 * W + x1;

  float s = 0.f;
  for (int c = 0; c < C; ++c) {
    const float* im = fc + (size_t)c * hw;
    const float v = w00 * im[i00] + w01 * im[i01] + w10 * im[i10] + w11 * im[i11];
    s += fabsf(fa[(size_t)c * hw + p] - v);
  }
  out[(size_t)d * hw + p] = s / (float)C;
}

// ---------------------------------------------------------------------------
// 2-channel bilinear resize with per-channel scale (flow / upfeat upsample)
// ---------------------------------------------------------------------------
__global__ void resize2_kernel(const float* __restrict__ in, int hi, int wi,
                               float* __restrict__ out, int ho, int wo,
                               float s0, float s1)
{
  const int idx = blockIdx.x * blockDim.x + threadIdx.x;
  const int hw = ho * wo;
  if (idx >= 2 * hw) return;
  const int c = idx / hw;
  const int p = idx - c * hw;
  const int py = p / wo, px = p - py * wo;
  const float sy = (py + 0.5f) * ((float)hi / (float)ho) - 0.5f;
  const float sx = (px + 0.5f) * ((float)wi / (float)wo) - 0.5f;
  const float v = bilerp_clamp(in + (size_t)c * hi * wi, hi, wi, sy, sx);
  out[idx] = v * (c == 0 ? s0 : s1);
}

// ---------------------------------------------------------------------------
// Driver: full PWC-Net forward. Concats done via channel-offset pointers.
// ---------------------------------------------------------------------------
static inline int cdiv(int a, int b) { return (a + b - 1) / b; }

extern "C" void kernel_launch(void* const* d_in, const int* in_sizes, int n_in,
                              void* d_out, int out_size, void* d_ws, size_t ws_size,
                              hipStream_t stream)
{
  (void)in_sizes; (void)n_in; (void)out_size; (void)ws_size;
  const float* im1 = (const float*)d_in[0];
  const float* im2 = (const float*)d_in[1];
  auto P = [&](int i) { return (const float*)d_in[i]; };

  float* ws = (float*)d_ws;
  size_t off = 0;
  auto alloc = [&](size_t n) { float* p = ws + off; off += n; return p; };

  // per-level geometry (index by pyramid level 2..6)
  const int CH[7] = {0, 0, 32, 64, 96, 128, 192};
  const int HH[7] = {0, 0, 64, 32, 16, 8, 4};
  const int WW[7] = {0, 0, 128, 64, 32, 16, 8};
  const int ODl[7] = {0, 0, 117, 149, 181, 213, 81};

  float* t0 = alloc((size_t)16 * 128 * 256);   // encoder ping-pong
  float* t1 = alloc((size_t)16 * 128 * 256);
  float* F[2][7];
  for (int im = 0; im < 2; ++im)
    for (int l = 2; l <= 6; ++l)
      F[im][l] = alloc((size_t)CH[l] * HH[l] * WW[l]);
  float* X[7];                                  // concat buffers (od+448 ch)
  for (int l = 2; l <= 6; ++l)
    X[l] = alloc((size_t)(ODl[l] + 448) * HH[l] * WW[l]);
  float* FL[7];                                 // flow per level
  for (int l = 2; l <= 6; ++l)
    FL[l] = alloc((size_t)2 * HH[l] * WW[l]);
  float* uft = alloc((size_t)2 * 32 * 64);      // upfeat temp (max: level-3)
  float* dA = alloc((size_t)128 * 64 * 128);    // context ping-pong
  float* dB = alloc((size_t)128 * 64 * 128);

  auto conv = [&](const float* in, int widx, float* out, const float* resid,
                  int Cin, int Cout, int Hin, int Win, int stride, int pad,
                  int dil, int relu, float oscale) {
    const int Hout = (Hin + 2 * pad - dil * 2 - 1) / stride + 1;
    const int Wout = (Win + 2 * pad - dil * 2 - 1) / stride + 1;
    if (Cout >= 32) {
      dim3 g(cdiv(Hout * Wout, TW), cdiv(Cout, 32));
      conv3x3_wmma<2><<<g, 256, 0, stream>>>(in, P(widx), P(widx + 1), out,
          resid, Cin, Cout, Hin, Win, Hout, Wout, stride, pad, dil, relu, oscale);
    } else {
      dim3 g(cdiv(Hout * Wout, TW), cdiv(Cout, 16));
      conv3x3_wmma<1><<<g, 256, 0, stream>>>(in, P(widx), P(widx + 1), out,
          resid, Cin, Cout, Hin, Win, Hout, Wout, stride, pad, dil, relu, oscale);
    }
  };

  // ---- encoder (params: idx 2.. : w,b per layer in ENC order) ----
  auto encode = [&](const float* im, float** Fr) {
    conv(im,     2, t0, nullptr,   3,  16, 256, 512, 2, 1, 1, 1, 1.f);
    conv(t0,     4, t1, nullptr,  16,  16, 128, 256, 1, 1, 1, 1, 1.f);
    conv(t1,     6, t0, nullptr,  16,  16, 128, 256, 1, 1, 1, 1, 1.f);
    conv(t0,     8, t1, nullptr,  16,  32, 128, 256, 2, 1, 1, 1, 1.f);
    conv(t1,    10, t0, nullptr,  32,  32,  64, 128, 1, 1, 1, 1, 1.f);
    conv(t0,    12, Fr[2], nullptr, 32, 32, 64, 128, 1, 1, 1, 1, 1.f);
    conv(Fr[2], 14, t0, nullptr,  32,  64,  64, 128, 2, 1, 1, 1, 1.f);
    conv(t0,    16, t1, nullptr,  64,  64,  32,  64, 1, 1, 1, 1, 1.f);
    conv(t1,    18, Fr[3], nullptr, 64, 64, 32,  64, 1, 1, 1, 1, 1.f);
    conv(Fr[3], 20, t0, nullptr,  64,  96,  32,  64, 2, 1, 1, 1, 1.f);
    conv(t0,    22, t1, nullptr,  96,  96,  16,  32, 1, 1, 1, 1, 1.f);
    conv(t1,    24, Fr[4], nullptr, 96, 96, 16,  32, 1, 1, 1, 1, 1.f);
    conv(Fr[4], 26, t0, nullptr,  96, 128,  16,  32, 2, 1, 1, 1, 1.f);
    conv(t0,    28, t1, nullptr, 128, 128,   8,  16, 1, 1, 1, 1, 1.f);
    conv(t1,    30, Fr[5], nullptr, 128, 128, 8, 16, 1, 1, 1, 1, 1.f);
    conv(Fr[5], 32, t0, nullptr, 128, 192,   8,  16, 2, 1, 1, 1, 1.f);
    conv(t0,    34, t1, nullptr, 192, 192,   4,   8, 1, 1, 1, 1, 1.f);
    conv(t1,    36, Fr[6], nullptr, 192, 192, 4,  8, 1, 1, 1, 1, 1.f);
  };
  encode(im1, F[0]);
  encode(im2, F[1]);

  auto warp = [&](int l, int sridx, const float* flow, int hf, int wf) {
    const int H = HH[l], W = WW[l];
    const int tot = 81 * H * W;
    dense_warp_kernel<<<cdiv(tot, 256), 256, 0, stream>>>(
        F[0][l], F[1][l], P(sridx), P(sridx + 1), flow, hf, wf,
        X[l] + (size_t)448 * H * W, CH[l], H, W);
  };

  // dense block: x = concat([y, x]) implemented by writing y just below x.
  auto dense = [&](int l, int widx) {
    const int H = HH[l], W = WW[l], npix = H * W, od = ODl[l];
    const int DDk[5] = {0, 128, 256, 352, 416};
    const int OUTk[5] = {128, 128, 96, 64, 32};
    int coff = 448;
    for (int k = 0; k < 5; ++k) {
      const int noff = coff - OUTk[k];
      conv(X[l] + (size_t)coff * npix, widx + 2 * k,
           X[l] + (size_t)noff * npix, nullptr,
           od + DDk[k], OUTk[k], H, W, 1, 1, 1, 1, 1.f);
      coff = noff;
    }
  };

  // ---- level 6 ----
  warp(6, 120, nullptr, 0, 0);
  dense(6, 38);
  conv(X[6], 48, FL[6], nullptr, ODl[6] + 448, 2, 4, 8, 1, 1, 1, 0, 20.f);

  // ---- levels 5..2 ----
  const int DB[7] = {0, 0, 94, 80, 66, 52, 0};     // dense-block base widx
  const int PF[7] = {0, 0, 104, 90, 76, 62, 48};   // predict_flow widx
  const int UF[7] = {0, 0, 92, 78, 64, 50, 0};     // upfeat{l+1} widx
  const int SR[7] = {0, 0, 128, 126, 124, 122, 120};

  for (int l = 5; l >= 2; --l) {
    const int H = HH[l], W = WW[l], npix = H * W;
    const int hp = HH[l + 1], wp = WW[l + 1];
    const int Cl = CH[l], od = ODl[l];

    // up_flow * DIV_FLOW -> channels [529+Cl, 531+Cl)
    {
      const float s0 = ((float)W / (float)wp) * 0.05f;
      const float s1 = ((float)H / (float)hp) * 0.05f;
      resize2_kernel<<<cdiv(2 * npix, 256), 256, 0, stream>>>(
          FL[l + 1], hp, wp, X[l] + (size_t)(529 + Cl) * npix, H, W, s0, s1);
    }
    // up_feat = resize(leaky(conv(upfeat{l+1}, X[l+1]))) -> channels [531+Cl, 533+Cl)
    conv(X[l + 1], UF[l], uft, nullptr, ODl[l + 1] + 448, 2, hp, wp, 1, 1, 1, 1, 1.f);
    resize2_kernel<<<cdiv(2 * npix, 256), 256, 0, stream>>>(
        uft, hp, wp, X[l] + (size_t)(531 + Cl) * npix, H, W, 1.f, 1.f);
    // corr -> channels [448, 529)
    warp(l, SR[l], FL[l + 1], hp, wp);
    // f1[l] -> channels [529, 529+Cl)
    hipMemcpyAsync(X[l] + (size_t)529 * npix, F[0][l],
                   (size_t)Cl * npix * sizeof(float),
                   hipMemcpyDeviceToDevice, stream);
    dense(l, DB[l]);
    conv(X[l], PF[l], FL[l], nullptr, od + 448, 2, H, W, 1, 1, 1, 0, 20.f);
  }

  // ---- context network (dilated convs) + residual flow refinement ----
  conv(X[2], 106, dA, nullptr, 565, 128, 64, 128, 1,  1,  1, 1, 1.f);
  conv(dA,   108, dB, nullptr, 128, 128, 64, 128, 1,  2,  2, 1, 1.f);
  conv(dB,   110, dA, nullptr, 128, 128, 64, 128, 1,  4,  4, 1, 1.f);
  conv(dA,   112, dB, nullptr, 128,  96, 64, 128, 1,  8,  8, 1, 1.f);
  conv(dB,   114, dA, nullptr,  96,  64, 64, 128, 1, 16, 16, 1, 1.f);
  conv(dA,   116, dB, nullptr,  64,  32, 64, 128, 1,  1,  1, 1, 1.f);
  conv(dB,   118, (float*)d_out, FL[2], 32, 2, 64, 128, 1, 1, 1, 0, 1.f);
}
